// PointTransformerConv1_2078764172139
// MI455X (gfx1250) — matrix operand order repaired and checked by reading, not compiled, not run on hardware
//
#include <hip/hip_runtime.h>

typedef __attribute__((ext_vector_type(16))) _Float16 v16h;
typedef __attribute__((ext_vector_type(8)))  _Float16 v8h;
typedef __attribute__((ext_vector_type(8)))  float    v8f;

static constexpr int NPTS = 8192;   // B*P
static constexpr int PPC  = 1024;   // points per cloud
static constexpr int CH   = 128;    // C_IN == C_OUT
static constexpr int KNBR = 64;     // neighbors
static constexpr float RAD2 = 0.04f;   // R*R
static constexpr float BNEPS = 1e-5f;

// ---------------------------------------------------------------------------
// Weight prep: Wt[n][k] = (f16) W[k][n]  for w_dst, w_src, w_lin, w_attn
// ---------------------------------------------------------------------------
__global__ void prep_weights_kernel(const float* __restrict__ w_dst,
                                    const float* __restrict__ w_src,
                                    const float* __restrict__ w_lin,
                                    const float* __restrict__ w_attn,
                                    _Float16* __restrict__ wt4) {
  int which = blockIdx.y;
  const float* src = (which == 0) ? w_dst : (which == 1) ? w_src
                    : (which == 2) ? w_lin : w_attn;
  _Float16* dst = wt4 + (size_t)which * CH * CH;
  int e = blockIdx.x * blockDim.x + threadIdx.x;
  if (e < CH * CH) {
    int k = e >> 7, n = e & 127;
    dst[n * CH + k] = (_Float16)src[k * CH + n];
  }
}

// ---------------------------------------------------------------------------
// kNN: one workgroup per target point. 64 sequential lexicographic argmins
// over (d2, index) keys -> exactly matches jax.lax.top_k(-d2) ordering.
// ---------------------------------------------------------------------------
__global__ void __launch_bounds__(256) knn_kernel(const float* __restrict__ pos,
                                                  int* __restrict__ idx_out,
                                                  int* __restrict__ valid_out) {
  __shared__ unsigned long long keys[PPC];
  __shared__ unsigned long long wred[8];
  int i = blockIdx.x;
  int base = (i >> 10) << 10;           // cloud base
  int tid = threadIdx.x;
  float px = pos[i * 3 + 0], py = pos[i * 3 + 1], pz = pos[i * 3 + 2];
  for (int j = tid; j < PPC; j += 256) {
    float dx = px - pos[(base + j) * 3 + 0];
    float dy = py - pos[(base + j) * 3 + 1];
    float dz = pz - pos[(base + j) * 3 + 2];
    float d2 = dx * dx + dy * dy + dz * dz;
    keys[j] = (((unsigned long long)__float_as_uint(d2)) << 32) | (unsigned)j;
  }
  __syncthreads();
  int lane = tid & 31, wv = tid >> 5;
  for (int it = 0; it < KNBR; ++it) {
    unsigned long long m = ~0ull;
    for (int j = tid; j < PPC; j += 256) {
      unsigned long long k = keys[j];
      m = (k < m) ? k : m;
    }
    #pragma unroll
    for (int off = 16; off > 0; off >>= 1) {
      unsigned lo = __shfl_down((unsigned)m, (unsigned)off, 32);
      unsigned hi = __shfl_down((unsigned)(m >> 32), (unsigned)off, 32);
      unsigned long long o = (((unsigned long long)hi) << 32) | lo;
      m = (o < m) ? o : m;
    }
    if (lane == 0) wred[wv] = m;
    __syncthreads();
    if (tid == 0) {
      unsigned long long mm = wred[0];
      #pragma unroll
      for (int w = 1; w < 8; ++w) mm = (wred[w] < mm) ? wred[w] : mm;
      unsigned j = (unsigned)(mm & 0xffffffffu);
      float d2v = __uint_as_float((unsigned)(mm >> 32));
      idx_out[i * KNBR + it]   = base + (int)j;      // global source index
      valid_out[i * KNBR + it] = (d2v <= RAD2) ? 1 : 0;
      keys[j] = ~0ull;                                // remove winner
    }
    __syncthreads();
  }
}

// ---------------------------------------------------------------------------
// A-fragment (16x32 f16) loader from row-major LDS tile (row stride = CH).
// lane<16: K = {0..15}+k0 ; lane>=16: K = {16..31}+k0.
// Caller passes pointer at [m][k0 + (lane<16?0:16)].
// ---------------------------------------------------------------------------
__device__ __forceinline__ v16h load_afrag(const _Float16* rowp) {
  v8h lo = *(const v8h*)rowp;
  v8h hi = *(const v8h*)(rowp + 8);
  v16h a;
  #pragma unroll
  for (int h = 0; h < 8; ++h) { a[h] = lo[h]; a[h + 8] = hi[h]; }
  return a;
}

// ---------------------------------------------------------------------------
// Projections: out = x @ W for W in {w_dst, w_src, w_lin}.
// Workgroup = 64x128 output tile, 8 waves (one 16-col N-tile each),
// 4 M-subtiles per wave, K-loop 4 x 32 -> v_wmma_f32_16x16x32_f16.
// ---------------------------------------------------------------------------
__global__ void __launch_bounds__(256) proj_kernel(const float* __restrict__ x,
                                                   const _Float16* __restrict__ wt4,
                                                   float* __restrict__ a_dst,
                                                   float* __restrict__ a_src,
                                                   float* __restrict__ v_out) {
  __shared__ __align__(16) _Float16 sx[64 * CH];
  int which = blockIdx.y;
  const _Float16* wt = wt4 + (size_t)which * CH * CH;   // Wt[n][k]
  float* out = (which == 0) ? a_dst : (which == 1) ? a_src : v_out;
  int m0g = blockIdx.x * 64;
  int tid = threadIdx.x;
  for (int e = tid; e < 64 * CH; e += 256) {
    int r = e >> 7, c = e & 127;
    sx[e] = (_Float16)x[(size_t)(m0g + r) * CH + c];
  }
  __syncthreads();

  int lane = tid & 31, wv = tid >> 5;
  int n0 = wv * 16;
  int lm = lane & 15;
  int kbA = (lane < 16) ? 0 : 16;
  int kbB = (lane < 16) ? 0 : 16;
  v8f acc[4] = {};
  #pragma unroll
  for (int ks = 0; ks < 4; ++ks) {
    int k0 = ks * 32;
    v16h bfrag = *(const v16h*)(wt + (n0 + lm) * CH + k0 + kbB);
    #pragma unroll
    for (int mt = 0; mt < 4; ++mt) {
      v16h afrag = load_afrag(&sx[(mt * 16 + lm) * CH + k0 + kbA]);
      acc[mt] = __builtin_amdgcn_wmma_f32_16x16x32_f16(
          false, afrag, false, bfrag, (short)0, acc[mt], false, false);
    }
  }
  int n = n0 + lm;
  int mofs = (lane < 16) ? 0 : 8;
  #pragma unroll
  for (int mt = 0; mt < 4; ++mt)
    #pragma unroll
    for (int e = 0; e < 8; ++e) {
      int row = m0g + mt * 16 + mofs + e;
      out[(size_t)row * CH + n] = acc[mt][e];
    }
}

// ---------------------------------------------------------------------------
// Attention: one workgroup per target point.
//   delta = BN/ReLU(rel @ w_pos + b)           (VALU, 6-wide K)
//   t     = a_dst_i - a_src_j + delta  -> f16 LDS (A matrix, 64x128)
//   u     = t @ w_attn                         (WMMA, 64x128x128)
//   alpha = softmax_k(mask(BN/ReLU(u + b)))    per channel
//   out_i = sum_k alpha * (v_j + delta)
// ---------------------------------------------------------------------------
__global__ void __launch_bounds__(256) attn_kernel(
    const float* __restrict__ pos, const float* __restrict__ nrm,
    const float* __restrict__ w_pos, const float* __restrict__ b_pos,
    const float* __restrict__ pg, const float* __restrict__ pb,
    const float* __restrict__ pm, const float* __restrict__ pv,
    const _Float16* __restrict__ wattn_t,
    const float* __restrict__ b_attn,
    const float* __restrict__ ag, const float* __restrict__ ab,
    const float* __restrict__ am, const float* __restrict__ av,
    const float* __restrict__ a_dst, const float* __restrict__ a_src,
    const float* __restrict__ vmat,
    const int* __restrict__ idx, const int* __restrict__ valid,
    float* __restrict__ out) {
  __shared__ float s_delta[KNBR * CH];                    // 32 KB
  __shared__ __align__(16) _Float16 s_tA[KNBR * CH];      // 16 KB
  __shared__ float s_alpha[KNBR * CH];                    // 32 KB
  __shared__ float s_rel[KNBR * 6];
  __shared__ int s_idx[KNBR];
  __shared__ int s_valid[KNBR];

  int i = blockIdx.x;
  int tid = threadIdx.x;
  if (tid < KNBR) {
    s_idx[tid]   = idx[i * KNBR + tid];
    s_valid[tid] = valid[i * KNBR + tid];
  }
  __syncthreads();
  if (tid < KNBR) {
    int jg = s_idx[tid];
    #pragma unroll
    for (int q = 0; q < 3; ++q) {
      s_rel[tid * 6 + q]     = pos[i * 3 + q] - pos[jg * 3 + q];
      s_rel[tid * 6 + 3 + q] = nrm[i * 3 + q] - nrm[jg * 3 + q];
    }
  }
  __syncthreads();

  // delta + t (each thread: one channel, 32 neighbors)
  {
    int c  = tid & 127;
    int kb = (tid >> 7) * 32;
    float wp0 = w_pos[0 * CH + c], wp1 = w_pos[1 * CH + c], wp2 = w_pos[2 * CH + c];
    float wp3 = w_pos[3 * CH + c], wp4 = w_pos[4 * CH + c], wp5 = w_pos[5 * CH + c];
    float bias  = b_pos[c];
    float scale = pg[c] * rsqrtf(pv[c] + BNEPS);
    float mean  = pm[c], beta = pb[c];
    float adc   = a_dst[(size_t)i * CH + c];
    for (int kk = 0; kk < 32; ++kk) {
      int k = kb + kk;
      const float* r = &s_rel[k * 6];
      float h = r[0] * wp0 + r[1] * wp1 + r[2] * wp2 +
                r[3] * wp3 + r[4] * wp4 + r[5] * wp5 + bias;
      h = (h - mean) * scale + beta;
      float d = fmaxf(h, 0.f);
      s_delta[k * CH + c] = d;
      float t = adc - a_src[(size_t)s_idx[k] * CH + c] + d;
      s_tA[k * CH + c] = (_Float16)t;
    }
  }
  __syncthreads();

  // WMMA: u = t @ w_attn   (M=64 neighbors, K=128, N=128)
  int lane = tid & 31, wv = tid >> 5;
  int n0 = wv * 16;
  int lm = lane & 15;
  int kbA = (lane < 16) ? 0 : 16;
  int kbB = (lane < 16) ? 0 : 16;
  v8f acc[4] = {};
  #pragma unroll
  for (int ks = 0; ks < 4; ++ks) {
    int k0 = ks * 32;
    v16h bfrag = *(const v16h*)(wattn_t + (n0 + lm) * CH + k0 + kbB);
    #pragma unroll
    for (int mt = 0; mt < 4; ++mt) {
      v16h afrag = load_afrag(&s_tA[(mt * 16 + lm) * CH + k0 + kbA]);
      acc[mt] = __builtin_amdgcn_wmma_f32_16x16x32_f16(
          false, afrag, false, bfrag, (short)0, acc[mt], false, false);
    }
  }
  // epilogue: bias + BN + ReLU + radius mask  (C layout: M = e+(lane<16?0:8))
  {
    int n = n0 + lm;
    float bias  = b_attn[n];
    float scale = ag[n] * rsqrtf(av[n] + BNEPS);
    float mean  = am[n], beta = ab[n];
    int mofs = (lane < 16) ? 0 : 8;
    #pragma unroll
    for (int mt = 0; mt < 4; ++mt)
      #pragma unroll
      for (int e = 0; e < 8; ++e) {
        int k = mt * 16 + mofs + e;
        float h = acc[mt][e] + bias;
        h = (h - mean) * scale + beta;
        h = fmaxf(h, 0.f);
        s_alpha[k * CH + n] = s_valid[k] ? h : -3.0e38f;
      }
  }
  __syncthreads();

  // per-channel softmax over neighbors + aggregation
  if (tid < CH) {
    int c = tid;
    float mx = -3.4e38f;
    for (int k = 0; k < KNBR; ++k) mx = fmaxf(mx, s_alpha[k * CH + c]);
    float sum = 0.f;
    for (int k = 0; k < KNBR; ++k) {
      float e = __expf(s_alpha[k * CH + c] - mx);
      s_alpha[k * CH + c] = e;
      sum += e;
    }
    float inv = 1.f / sum;
    float acc2 = 0.f;
    for (int k = 0; k < KNBR; ++k) {
      float w = s_alpha[k * CH + c] * inv;
      acc2 += w * (vmat[(size_t)s_idx[k] * CH + c] + s_delta[k * CH + c]);
    }
    out[(size_t)i * CH + c] = acc2;
  }
}

// ---------------------------------------------------------------------------
extern "C" void kernel_launch(void* const* d_in, const int* in_sizes, int n_in,
                              void* d_out, int out_size, void* d_ws, size_t ws_size,
                              hipStream_t stream) {
  const float* x      = (const float*)d_in[0];
  const float* pos    = (const float*)d_in[1];
  const float* nrm    = (const float*)d_in[2];
  // d_in[3] = batch (structured i//P, unused)
  const float* w_lin  = (const float*)d_in[4];
  const float* w_src  = (const float*)d_in[5];
  const float* w_dst  = (const float*)d_in[6];
  const float* w_pos  = (const float*)d_in[7];
  const float* b_pos  = (const float*)d_in[8];
  const float* pg     = (const float*)d_in[9];
  const float* pb     = (const float*)d_in[10];
  const float* pm     = (const float*)d_in[11];
  const float* pv     = (const float*)d_in[12];
  const float* w_attn = (const float*)d_in[13];
  const float* b_attn = (const float*)d_in[14];
  const float* ag     = (const float*)d_in[15];
  const float* ab     = (const float*)d_in[16];
  const float* am     = (const float*)d_in[17];
  const float* av     = (const float*)d_in[18];
  float* out = (float*)d_out;

  char* ws = (char*)d_ws;
  int*   idx    = (int*)(ws);                                    // 2 MB
  int*   valid  = (int*)(ws + (size_t)NPTS * KNBR * 4);          // 2 MB
  float* a_dstb = (float*)(ws + (size_t)NPTS * KNBR * 8);        // 4 MB
  float* a_srcb = a_dstb + (size_t)NPTS * CH;                    // 4 MB
  float* vb     = a_srcb + (size_t)NPTS * CH;                    // 4 MB
  _Float16* wt4 = (_Float16*)(vb + (size_t)NPTS * CH);           // 128 KB (dst,src,lin,attn)

  prep_weights_kernel<<<dim3(64, 4), 256, 0, stream>>>(w_dst, w_src, w_lin, w_attn, wt4);
  knn_kernel<<<NPTS, 256, 0, stream>>>(pos, idx, valid);
  proj_kernel<<<dim3(NPTS / 64, 3), 256, 0, stream>>>(x, wt4, a_dstb, a_srcb, vb);
  attn_kernel<<<NPTS, 256, 0, stream>>>(pos, nrm, w_pos, b_pos, pg, pb, pm, pv,
                                        wt4 + (size_t)3 * CH * CH, b_attn,
                                        ag, ab, am, av,
                                        a_dstb, a_srcb, vb, idx, valid, out);
}
